// NEDMP_layer_48962627174426
// MI455X (gfx1250) — matrix operand order repaired and checked by reading, not compiled, not run on hardware
//
#include <hip/hip_runtime.h>
#include <hip/hip_bf16.h>

// ---------------------------------------------------------------------------
// NEDMP layer for MI455X (gfx1250).  Dominant cost: streaming the 400MB
// nb_matrix (+80MB edge2tnode) once -> HBM-bound (~21us @ 23.3TB/s).
// Heavy GEMMs use v_wmma_f32_16x16x32_bf16 (adjacency 0/1 is exact in bf16),
// fp32 accumulate.  K is split across waves for latency hiding; partials are
// reduced deterministically (no float atomics).  Hot loop is branch-free;
// the single ragged K-block (K%32==16) is peeled out.
// ---------------------------------------------------------------------------

#define E_ 10000
#define N_ 2000
#define H_ 32
#define KBLK ((E_ + 31) / 32)   // 313 K-blocks of 32 (last block half-valid)

typedef __attribute__((ext_vector_type(16))) __bf16 v16bf;
typedef __attribute__((ext_vector_type(8)))  float  v8f;

__device__ __forceinline__ unsigned short f2bf(float f) {
  unsigned int u = __float_as_uint(f);
  unsigned int r = u + 0x7FFFu + ((u >> 16) & 1u);   // round-to-nearest-even
  return (unsigned short)(r >> 16);
}
// pack two fp32 -> two bf16 in one dword by truncation (EXACT for 0.0/1.0)
__device__ __forceinline__ unsigned int pack_trunc(float lo, float hi) {
  return (__float_as_uint(hi) & 0xFFFF0000u) | (__float_as_uint(lo) >> 16);
}
__device__ __forceinline__ float sigmoidf_(float x) {
  return 1.0f / (1.0f + __expf(-x));
}

// --------------------------- small prep kernels ----------------------------

__global__ void k_theta_em(const float* __restrict__ theta,
                           const float* __restrict__ W_theta, const float* __restrict__ b_theta,
                           const float* __restrict__ edge_message,
                           const float* __restrict__ W_m2, const float* __restrict__ b_m2,
                           float* __restrict__ theta_emb, float* __restrict__ em) {
  int t = blockIdx.x * blockDim.x + threadIdx.x;
  if (t >= E_ * H_) return;
  int e = t >> 5, h = t & 31;
  float a = b_theta[h] + theta[e * 3 + 0] * W_theta[0 * H_ + h]
                       + theta[e * 3 + 1] * W_theta[1 * H_ + h]
                       + theta[e * 3 + 2] * W_theta[2 * H_ + h];
  theta_emb[t] = fmaxf(a, 0.0f);
  em[t] = fmaxf(edge_message[e] * W_m2[h] + b_m2[h], 0.0f);
}

__global__ void k_nm(const float* __restrict__ node_message,
                     const float* __restrict__ W_m1, const float* __restrict__ b_m1,
                     float* __restrict__ nm) {
  int t = blockIdx.x * blockDim.x + threadIdx.x;
  if (t >= N_ * H_) return;
  int n = t >> 5, h = t & 31;
  nm[t] = fmaxf(node_message[n] * W_m1[h] + b_m1[h], 0.0f);
}

// h_cat = relu([hidden | theta_emb] @ W_cat + b_cat); scatter into WMMA
// B-operand layout (bf16): elem j of lane L (half selects N block):
//   K = kb*32 + (j<8 ? j : j+8) + 8*(L>>4),  N = 16*half + (L&15)
__global__ void k_hcat(const float* __restrict__ hidden,
                       const float* __restrict__ theta_emb,
                       const float* __restrict__ W_cat, const float* __restrict__ b_cat,
                       unsigned short* __restrict__ HcB) {
  int t = blockIdx.x * blockDim.x + threadIdx.x;
  if (t >= E_ * H_) return;
  int e = t >> 5, h = t & 31;
  float acc = b_cat[h];
  const float* hr = hidden + e * H_;
  const float* te = theta_emb + e * H_;
#pragma unroll
  for (int k = 0; k < H_; ++k) acc += hr[k] * W_cat[k * H_ + h];
#pragma unroll
  for (int k = 0; k < H_; ++k) acc += te[k] * W_cat[(H_ + k) * H_ + h];
  acc = fmaxf(acc, 0.0f);
  int kb = e >> 5, within = e & 31;
  int laneHi = (within >> 3) & 1, j = (within & 7) + ((within >> 4) << 3);
  int half = h >> 4, lane = laneHi * 16 + (h & 15);
  size_t idx = (((size_t)kb * 2 + half) * 32 + lane) * 16 + j;
  HcB[idx] = f2bf(acc);
}

// ------------------------------ WMMA GEMM ----------------------------------
// P_part[split] = A[tile rows, K-range] @ Hc   (A fp32 0/1 -> bf16 exact)
// One wave per (M-tile, K-split).  Block = 4 waves.
__global__ void k_gemm_wmma(const float* __restrict__ A,
                            const unsigned short* __restrict__ HcB,
                            float* __restrict__ Ppart,
                            int M, int K, int ksplit) {
  const int lane = threadIdx.x & 31;
  const int wv   = (blockIdx.x * blockDim.x + threadIdx.x) >> 5;
  const int tiles = M >> 4;
  const int split = wv % ksplit;
  const int tile  = wv / ksplit;
  if (tile >= tiles) return;

  const int sel  = lane >> 4;                 // A K-half select
  const int mrow = tile * 16 + (lane & 15);   // A row for this lane
  const float* arow = A + (size_t)mrow * K;

  const int kbtot  = (K + 31) / 32;           // 313
  const int kbFull = K / 32;                  // 312 (full 32-wide blocks)
  const int len = (kbtot + ksplit - 1) / ksplit;
  const int kb0 = split * len;
  int kb1 = kb0 + len; if (kb1 > kbtot) kb1 = kbtot;
  int kbMainEnd = kb1 < kbFull ? kb1 : kbFull;

  v8f c0 = {}; v8f c1 = {};
  union AV { unsigned int u32[8]; unsigned short u[16]; v16bf v; };

  // ---- branch-free hot loop: full 32-wide K-blocks ----
  for (int kb = kb0; kb < kbMainEnd; ++kb) {
    const float* ab = arow + kb * 32 + sel * 8;
    __builtin_prefetch(ab + 32 * 4, 0, 1);    // global_prefetch_b8, 4 kb ahead
    AV a;
    // A layout: elem j -> K-offset (j<8 ? j : j+8) + 8*sel.
    // Pairs are contiguous -> dword packs; truncation exact for 0/1 values.
#pragma unroll
    for (int p = 0; p < 8; ++p) {
      int koff = (p < 4) ? (2 * p) : (2 * p + 8);
      float2 f = *(const float2*)(ab + koff);
      a.u32[p] = pack_trunc(f.x, f.y);
    }
    // B: two coalesced 32B vector loads (layout pre-baked in ws, L2-resident)
    const unsigned short* bbase = HcB + (((size_t)kb * 2) * 32 + lane) * 16;
    v16bf b0 = *(const v16bf*)(bbase);
    v16bf b1 = *(const v16bf*)(bbase + 32 * 16);
    c0 = __builtin_amdgcn_wmma_f32_16x16x32_bf16(false, a.v, false, b0,
                                                 (short)0, c0, false, false);
    c1 = __builtin_amdgcn_wmma_f32_16x16x32_bf16(false, a.v, false, b1,
                                                 (short)0, c1, false, false);
  }

  // ---- peeled tail block (only the split owning kbFull runs this; K%32==16
  //      valid columns land exactly in elems j<8 per lane) ----
  if (kb1 > kbFull) {
    const float* ab = arow + kbFull * 32 + sel * 8;
    AV a;
#pragma unroll
    for (int p = 0; p < 4; ++p) {
      float2 f = *(const float2*)(ab + 2 * p);
      a.u32[p] = pack_trunc(f.x, f.y);
    }
#pragma unroll
    for (int p = 4; p < 8; ++p) a.u32[p] = 0;
    const unsigned short* bbase = HcB + (((size_t)kbFull * 2) * 32 + lane) * 16;
    v16bf b0 = *(const v16bf*)(bbase);
    v16bf b1 = *(const v16bf*)(bbase + 32 * 16);
    c0 = __builtin_amdgcn_wmma_f32_16x16x32_bf16(false, a.v, false, b0,
                                                 (short)0, c0, false, false);
    c1 = __builtin_amdgcn_wmma_f32_16x16x32_bf16(false, a.v, false, b1,
                                                 (short)0, c1, false, false);
  }

  // D layout: vgpr r -> m = r + 8*sel (within tile), n = lane&15 (+16 for c1)
  float* Pp = Ppart + (size_t)split * M * H_;
#pragma unroll
  for (int r = 0; r < 8; ++r) {
    int m = tile * 16 + r + 8 * sel;
    Pp[(size_t)m * H_ + (lane & 15)]      = c0[r];
    Pp[(size_t)m * H_ + 16 + (lane & 15)] = c1[r];
  }
}

// Deterministic reduction of K-split partials
__global__ void k_reduce(const float* __restrict__ Ppart, float* __restrict__ P,
                         int M32, int ksplit) {
  int t = blockIdx.x * blockDim.x + threadIdx.x;
  if (t >= M32) return;
  float s = 0.0f;
  for (int i = 0; i < ksplit; ++i) s += Ppart[(size_t)i * M32 + t];
  P[t] = s;
}

// out[m,h] = relu(P[m,:] @ W + b)   (W_agg1 / W_agg2 epilogue)
__global__ void k_agg(const float* __restrict__ P, const float* __restrict__ W,
                      const float* __restrict__ b, float* __restrict__ out, int M) {
  int t = blockIdx.x * blockDim.x + threadIdx.x;
  if (t >= M * H_) return;
  int m = t >> 5, h = t & 31;
  float acc = b[h];
  const float* pr = P + (size_t)m * H_;
#pragma unroll
  for (int k = 0; k < H_; ++k) acc += pr[k] * W[k * H_ + h];
  out[t] = fmaxf(acc, 0.0f);
}

// node_res = sigmoid([node_agg | nm] @ W_sd1 + b_sd1) -> out[0:N], out[N:2N]
__global__ void k_node_res(const float* __restrict__ node_agg, const float* __restrict__ nm,
                           const float* __restrict__ W_sd1, const float* __restrict__ b_sd1,
                           float* __restrict__ out) {
  int n = blockIdx.x * blockDim.x + threadIdx.x;
  if (n >= N_) return;
  float s0 = b_sd1[0], s1 = b_sd1[1];
  const float* na = node_agg + (size_t)n * H_;
  const float* nmp = nm + (size_t)n * H_;
#pragma unroll
  for (int k = 0; k < H_; ++k) { s0 += na[k] * W_sd1[k * 2]; s1 += na[k] * W_sd1[k * 2 + 1]; }
#pragma unroll
  for (int k = 0; k < H_; ++k) { s0 += nmp[k] * W_sd1[(H_ + k) * 2]; s1 += nmp[k] * W_sd1[(H_ + k) * 2 + 1]; }
  out[n]      = sigmoidf_(s0);
  out[N_ + n] = sigmoidf_(s1);
}

// edge_res = sigmoid([hidden_agg | em] @ W_sd2 + b_sd2) -> out[2N:2N+E], out[2N+E:2N+2E]
__global__ void k_edge_res(const float* __restrict__ hidden_agg, const float* __restrict__ em,
                           const float* __restrict__ W_sd2, const float* __restrict__ b_sd2,
                           float* __restrict__ out) {
  int e = blockIdx.x * blockDim.x + threadIdx.x;
  if (e >= E_) return;
  float s0 = b_sd2[0], s1 = b_sd2[1];
  const float* ha = hidden_agg + (size_t)e * H_;
  const float* emp = em + (size_t)e * H_;
#pragma unroll
  for (int k = 0; k < H_; ++k) { s0 += ha[k] * W_sd2[k * 2]; s1 += ha[k] * W_sd2[k * 2 + 1]; }
#pragma unroll
  for (int k = 0; k < H_; ++k) { s0 += emp[k] * W_sd2[(H_ + k) * 2]; s1 += emp[k] * W_sd2[(H_ + k) * 2 + 1]; }
  out[2 * N_ + e]      = sigmoidf_(s0);
  out[2 * N_ + E_ + e] = sigmoidf_(s1);
}

// GRU cell: x = hidden_agg, h = hidden -> new_hidden -> out[2N+2E : ]
__global__ void k_gru(const float* __restrict__ hidden_agg, const float* __restrict__ hidden,
                      const float* __restrict__ W_ih, const float* __restrict__ b_ih,
                      const float* __restrict__ W_hh, const float* __restrict__ b_hh,
                      float* __restrict__ out) {
  int t = blockIdx.x * blockDim.x + threadIdx.x;
  if (t >= E_ * H_) return;
  int e = t >> 5, h = t & 31;
  const float* xa = hidden_agg + (size_t)e * H_;
  const float* hs = hidden + (size_t)e * H_;
  float ir = b_ih[h], iz = b_ih[H_ + h], in_ = b_ih[2 * H_ + h];
  float hr = b_hh[h], hz = b_hh[H_ + h], hn = b_hh[2 * H_ + h];
#pragma unroll
  for (int k = 0; k < H_; ++k) {
    float x = xa[k], hh = hs[k];
    const float* wi = W_ih + k * 3 * H_;
    const float* wh = W_hh + k * 3 * H_;
    ir += x * wi[h];      iz += x * wi[H_ + h];      in_ += x * wi[2 * H_ + h];
    hr += hh * wh[h];     hz += hh * wh[H_ + h];     hn += hh * wh[2 * H_ + h];
  }
  float r = sigmoidf_(ir + hr);
  float z = sigmoidf_(iz + hz);
  float n = tanhf(in_ + r * hn);
  out[2 * N_ + 2 * E_ + t] = (1.0f - z) * n + z * hs[h];
}

// ------------------------------- launcher ----------------------------------

extern "C" void kernel_launch(void* const* d_in, const int* in_sizes, int n_in,
                              void* d_out, int out_size, void* d_ws, size_t ws_size,
                              hipStream_t stream) {
  const float* theta        = (const float*)d_in[0];
  const float* edge_message = (const float*)d_in[1];
  const float* node_message = (const float*)d_in[2];
  const float* nb_matrix    = (const float*)d_in[3];
  const float* edge2tnode   = (const float*)d_in[4];
  const float* hidden       = (const float*)d_in[5];
  const float* W_theta = (const float*)d_in[6];  const float* b_theta = (const float*)d_in[7];
  const float* W_m1    = (const float*)d_in[8];  const float* b_m1    = (const float*)d_in[9];
  const float* W_m2    = (const float*)d_in[10]; const float* b_m2    = (const float*)d_in[11];
  const float* W_agg1  = (const float*)d_in[12]; const float* b_agg1  = (const float*)d_in[13];
  const float* W_agg2  = (const float*)d_in[14]; const float* b_agg2  = (const float*)d_in[15];
  const float* W_cat   = (const float*)d_in[16]; const float* b_cat   = (const float*)d_in[17];
  const float* W_ih    = (const float*)d_in[18]; const float* b_ih    = (const float*)d_in[19];
  const float* W_hh    = (const float*)d_in[20]; const float* b_hh    = (const float*)d_in[21];
  const float* W_sd1   = (const float*)d_in[22]; const float* b_sd1   = (const float*)d_in[23];
  const float* W_sd2   = (const float*)d_in[24]; const float* b_sd2   = (const float*)d_in[25];
  float* out = (float*)d_out;
  (void)in_sizes; (void)n_in; (void)out_size;

  // ---- workspace layout (256B aligned slabs), ksplit chosen to fit ws ----
  const size_t SZ_EH  = (size_t)E_ * H_ * sizeof(float);   // 1.28 MB
  const size_t SZ_NH  = (size_t)N_ * H_ * sizeof(float);   // 256 KB
  const size_t SZ_HCB = (size_t)KBLK * 2 * 32 * 16 * sizeof(unsigned short); // ~641 KB
  auto al = [](size_t x) { return (x + 255) & ~(size_t)255; };

  int ksplit = 8;
  for (;;) {
    size_t need = al(SZ_EH) * 4 /*theta_emb, em, P_e, hidden_agg*/
                + al(SZ_NH) * 3 /*nm, P_n, node_agg*/
                + al(SZ_HCB)
                + al((size_t)ksplit * SZ_EH) + al((size_t)ksplit * SZ_NH);
    if (need <= ws_size || ksplit == 1) break;
    ksplit >>= 1;
  }

  char* w = (char*)d_ws;
  auto grab = [&](size_t bytes) { char* p = w; w += al(bytes); return p; };
  float*          theta_emb  = (float*)grab(SZ_EH);
  float*          em         = (float*)grab(SZ_EH);
  float*          nm         = (float*)grab(SZ_NH);
  unsigned short* HcB        = (unsigned short*)grab(SZ_HCB);
  float*          Ppart_e    = (float*)grab((size_t)ksplit * SZ_EH);
  float*          Ppart_n    = (float*)grab((size_t)ksplit * SZ_NH);
  float*          P_e        = (float*)grab(SZ_EH);
  float*          P_n        = (float*)grab(SZ_NH);
  float*          hidden_agg = (float*)grab(SZ_EH);
  float*          node_agg   = (float*)grab(SZ_NH);

  // Zero the B-layout buffer (tail K-block must be zero-padded)
  hipMemsetAsync(HcB, 0, SZ_HCB, stream);

  const int TB = 256;
  // 1) theta_emb + em ; nm
  k_theta_em<<<(E_ * H_ + TB - 1) / TB, TB, 0, stream>>>(theta, W_theta, b_theta,
                                                         edge_message, W_m2, b_m2,
                                                         theta_emb, em);
  k_nm<<<(N_ * H_ + TB - 1) / TB, TB, 0, stream>>>(node_message, W_m1, b_m1, nm);
  // 2) h_cat -> WMMA B layout (bf16)
  k_hcat<<<(E_ * H_ + TB - 1) / TB, TB, 0, stream>>>(hidden, theta_emb, W_cat, b_cat, HcB);

  // 3) the two big WMMA GEMMs (4 waves = 128 threads per block)
  {
    int tiles_e = E_ / 16, tiles_n = N_ / 16;
    int waves_e = tiles_e * ksplit, waves_n = tiles_n * ksplit;
    k_gemm_wmma<<<(waves_e + 3) / 4, 128, 0, stream>>>(nb_matrix, HcB, Ppart_e, E_, E_, ksplit);
    k_gemm_wmma<<<(waves_n + 3) / 4, 128, 0, stream>>>(edge2tnode, HcB, Ppart_n, N_, E_, ksplit);
  }
  // 4) deterministic K-split reductions
  k_reduce<<<(E_ * H_ + TB - 1) / TB, TB, 0, stream>>>(Ppart_e, P_e, E_ * H_, ksplit);
  k_reduce<<<(N_ * H_ + TB - 1) / TB, TB, 0, stream>>>(Ppart_n, P_n, N_ * H_, ksplit);

  // 5) agg epilogues
  k_agg<<<(N_ * H_ + TB - 1) / TB, TB, 0, stream>>>(P_n, W_agg1, b_agg1, node_agg, N_);
  k_agg<<<(E_ * H_ + TB - 1) / TB, TB, 0, stream>>>(P_e, W_agg2, b_agg2, hidden_agg, E_);

  // 6) heads + GRU
  k_node_res<<<(N_ + TB - 1) / TB, TB, 0, stream>>>(node_agg, nm, W_sd1, b_sd1, out);
  k_edge_res<<<(E_ + TB - 1) / TB, TB, 0, stream>>>(hidden_agg, em, W_sd2, b_sd2, out);
  k_gru<<<(E_ * H_ + TB - 1) / TB, TB, 0, stream>>>(hidden_agg, hidden,
                                                    W_ih, b_ih, W_hh, b_hh, out);
}
